// single_47974784696696
// MI455X (gfx1250) — matrix operand order repaired and checked by reading, not compiled, not run on hardware
//
#include <hip/hip_runtime.h>

// ---------------------------------------------------------------------------
// Problem constants (match the reference)
// ---------------------------------------------------------------------------
#define BB   1024
#define FEA  4096
#define HH   1024
#define TT   64
#define GAP  16

typedef __attribute__((ext_vector_type(16))) __bf16 v16bf;
typedef __attribute__((ext_vector_type(8)))  float  v8f;

union BF16Frag {
    v16bf v;
    uint4 q[2];
};

// round-to-nearest-even fp32 -> bf16 bits
__device__ __forceinline__ unsigned short f2bf(float f) {
    unsigned int x = __builtin_bit_cast(unsigned int, f);
    unsigned int r = (x + 0x7FFFu + ((x >> 16) & 1u)) >> 16;
    return (unsigned short)r;
}

// ---------------------------------------------------------------------------
// Conversion / prep kernels
// ---------------------------------------------------------------------------
__global__ __launch_bounds__(256) void k_f32_to_bf16(const float* __restrict__ in,
                                                     unsigned short* __restrict__ out, int n) {
    int i = blockIdx.x * blockDim.x + threadIdx.x;
    if (i < n) out[i] = f2bf(in[i]);
}

__global__ __launch_bounds__(256) void k_sum_to_bf16(const float* __restrict__ a,
                                                     const float* __restrict__ b,
                                                     unsigned short* __restrict__ out, int n) {
    int i = blockIdx.x * blockDim.x + threadIdx.x;
    if (i < n) out[i] = f2bf(a[i] + b[i]);
}

__global__ __launch_bounds__(256) void k_add_f32(const float* __restrict__ a,
                                                 const float* __restrict__ b,
                                                 float* __restrict__ out, int n) {
    int i = blockIdx.x * blockDim.x + threadIdx.x;
    if (i < n) out[i] = a[i] + b[i];
}

// Final fused sigmoid over all T*B logits: out[i] = sigmoid(sum_c parts[i*16+c] + ba)
__global__ __launch_bounds__(256) void k_sigmoid_all(const float* __restrict__ parts,
                                                     const float* __restrict__ ba,
                                                     float* __restrict__ out, int n, int chunks) {
    int i = blockIdx.x * blockDim.x + threadIdx.x;
    if (i >= n) return;
    float s = ba[0];
    const float* p = parts + (size_t)i * chunks;
    for (int c = 0; c < chunks; ++c) s += p[c];
    out[i] = 1.0f / (1.0f + __expf(-s));
}

// ---------------------------------------------------------------------------
// Fragment loads (this code's contiguous k order, same for A and B, wave32)
//
// A 16x32 bf16 (MxK): lane L (0..15)  holds row M=L,   K = 0..15  (contiguous)
//                     lane L (16..31) holds row M=L-16, K = 16..31 (contiguous)
// B 32x16 bf16 (KxN): lane L (0..15)  holds col N=L,    K = 0..15  (contiguous)
//                     lane L (16..31) holds col N=L-16, K = 16..31 (contiguous)
// out = act @ W^T with act[M,K], W[N,K] both row-major -> B columns are W rows,
// so both operands are contiguous per-lane loads.
// ---------------------------------------------------------------------------
__device__ __forceinline__ v16bf load_A_frag(const unsigned short* __restrict__ A, int lda,
                                             int row0, int k0, int lane) {
    int r  = row0 + (lane & 15);
    int kb = k0 + ((lane >> 4) << 4);  // +0 or +16
    const unsigned short* p = A + (size_t)r * lda + kb;
    BF16Frag f;
    f.q[0] = *(const uint4*)(p);        // K = kb .. kb+7
    f.q[1] = *(const uint4*)(p + 8);    // K = kb+8 .. kb+15
    return f.v;
}

__device__ __forceinline__ v16bf load_B_frag(const unsigned short* __restrict__ W, int ldw,
                                             int col0, int k0, int lane) {
    int n  = col0 + (lane & 15);
    int kb = k0 + ((lane >> 4) << 4);  // +0 or +16
    const unsigned short* p = W + (size_t)n * ldw + kb;
    BF16Frag f;
    f.q[0] = *(const uint4*)(p);        // K = kb .. kb+7
    f.q[1] = *(const uint4*)(p + 8);    // K = kb+8 .. kb+15
    return f.v;
}

// ---------------------------------------------------------------------------
// Split-K tiled WMMA GEMM:
//   out[M,N] = act( A1@B1^T (+ A2@B2^T) + bias )
//   A*: bf16 [M,K] row-major      B*: bf16 [N,K] row-major (i.e. W)
//   act: 0=none 1=relu 2=tanh
//
// Block = 128 threads = 4 waves = 2 output tiles x 2 K-halves.
// Each wave runs K/2 with a 64x64 tile (4x4 v8f accumulators, 16 WMMA/K-step
// per operand pair). The kHalf==1 wave dumps its accumulators to LDS; the
// kHalf==0 wave merges and runs the epilogue (bias + activation + bf16 store
// + fused per-row partial dot with Wa for the actor head).
// ---------------------------------------------------------------------------
__global__ __launch_bounds__(128) void k_gemm_wmma(
    const unsigned short* __restrict__ A1, const unsigned short* __restrict__ B1,
    const unsigned short* __restrict__ A2, const unsigned short* __restrict__ B2,
    const float* __restrict__ bias,
    float* __restrict__ outF, unsigned short* __restrict__ outBf,
    const float* __restrict__ Wa, float* __restrict__ probPart,
    int M, int N, int K, int actMode, int tilesN) {

    __shared__ float red[2][4096];  // 2 tiles x 64x64 f32 partials (32 KB)

    const int wid   = threadIdx.x >> 5;
    const int lane  = threadIdx.x & 31;
    const int tLoc  = wid >> 1;        // tile within block (0..1)
    const int kHalf = wid & 1;         // K partition (0..1)
    const int tile  = blockIdx.x * 2 + tLoc;
    const int tm = tile / tilesN;
    const int tn = tile % tilesN;
    const int m0 = tm * 64;
    const int n0 = tn * 64;

    v8f acc[4][4] = {};

    const int kh = K >> 1;             // per-wave K range (multiple of 32)
    const int k0 = kHalf * kh;
    const int k1 = k0 + kh;

    for (int k = k0; k < k1; k += 32) {
        v16bf a[4], b[4];
#pragma unroll
        for (int i = 0; i < 4; ++i) a[i] = load_A_frag(A1, K, m0 + 16 * i, k, lane);
#pragma unroll
        for (int j = 0; j < 4; ++j) b[j] = load_B_frag(B1, K, n0 + 16 * j, k, lane);
#pragma unroll
        for (int i = 0; i < 4; ++i)
#pragma unroll
            for (int j = 0; j < 4; ++j)
                acc[i][j] = __builtin_amdgcn_wmma_f32_16x16x32_bf16(
                    false, a[i], false, b[j], (short)0, acc[i][j], false, false);
    }

    if (A2 != nullptr) {  // dual-input steps (t = 16/32/48): + hx @ Whh^T
        for (int k = k0; k < k1; k += 32) {
            v16bf a[4], b[4];
#pragma unroll
            for (int i = 0; i < 4; ++i) a[i] = load_A_frag(A2, K, m0 + 16 * i, k, lane);
#pragma unroll
            for (int j = 0; j < 4; ++j) b[j] = load_B_frag(B2, K, n0 + 16 * j, k, lane);
#pragma unroll
            for (int i = 0; i < 4; ++i)
#pragma unroll
                for (int j = 0; j < 4; ++j)
                    acc[i][j] = __builtin_amdgcn_wmma_f32_16x16x32_bf16(
                        false, a[i], false, b[j], (short)0, acc[i][j], false, false);
        }
    }

    // ---- split-K merge through LDS ----
    if (kHalf == 1) {
#pragma unroll
        for (int i = 0; i < 4; ++i)
#pragma unroll
            for (int j = 0; j < 4; ++j)
                *(v8f*)&red[tLoc][(i * 4 + j) * 256 + lane * 8] = acc[i][j];
    }
    __syncthreads();
    if (kHalf == 1) return;

#pragma unroll
    for (int i = 0; i < 4; ++i)
#pragma unroll
        for (int j = 0; j < 4; ++j)
            acc[i][j] += *(const v8f*)&red[tLoc][(i * 4 + j) * 256 + lane * 8];

    // ---- epilogue ----
    // C/D 16x16 f32 layout: VGPR r -> lanes 0-15: M=r, lanes 16-31: M=8+r; N = lane%16
    const int ln = lane & 15;
    const int lh = lane >> 4;
    float bv4[4], wa4[4];
#pragma unroll
    for (int j = 0; j < 4; ++j) {
        const int n = n0 + 16 * j + ln;
        bv4[j] = bias ? bias[n] : 0.0f;
        wa4[j] = probPart ? Wa[n] : 0.0f;
    }
#pragma unroll
    for (int i = 0; i < 4; ++i) {
#pragma unroll
        for (int r = 0; r < 8; ++r) {
            const int m = m0 + 16 * i + r + 8 * lh;
            float pr = 0.0f;
#pragma unroll
            for (int j = 0; j < 4; ++j) {
                float v = acc[i][j][r] + bv4[j];
                if (actMode == 1) v = fmaxf(v, 0.0f);
                else if (actMode == 2) v = tanhf(v);
                const size_t off = (size_t)m * N + (n0 + 16 * j + ln);
                if (outF) outF[off] = v;
                outBf[off] = f2bf(v);
                pr += v * wa4[j];
            }
            if (probPart) {
                // reduce over the 16 lanes of this half-wave (bits 0..3 of lane id)
                pr += __shfl_xor(pr, 1, 32);
                pr += __shfl_xor(pr, 2, 32);
                pr += __shfl_xor(pr, 4, 32);
                pr += __shfl_xor(pr, 8, 32);
                if (ln == 0) probPart[(size_t)m * tilesN + tn] = pr;  // unique (m,tn) slot
            }
        }
    }
}

// ---------------------------------------------------------------------------
// Host-side orchestration
// ---------------------------------------------------------------------------
extern "C" void kernel_launch(void* const* d_in, const int* in_sizes, int n_in,
                              void* d_out, int out_size, void* d_ws, size_t ws_size,
                              hipStream_t stream) {
    const float* x   = (const float*)d_in[0];
    const float* W1  = (const float*)d_in[1];
    const float* b1  = (const float*)d_in[2];
    const float* W2  = (const float*)d_in[3];
    const float* b2  = (const float*)d_in[4];
    const float* Wih = (const float*)d_in[5];
    const float* Whh = (const float*)d_in[6];
    const float* bih = (const float*)d_in[7];
    const float* bhh = (const float*)d_in[8];
    const float* Wa  = (const float*)d_in[9];
    const float* ba  = (const float*)d_in[10];
    float* out = (float*)d_out;  // [T*B probs][B*H hx]

    // workspace carving (256B aligned)
    char* ws = (char*)d_ws;
    size_t cur = 0;
    auto carve = [&](size_t bytes) {
        void* p = ws + cur;
        cur += (bytes + 255) & ~(size_t)255;
        return p;
    };
    unsigned short* Xb    = (unsigned short*)carve((size_t)BB * FEA * 2);
    unsigned short* W1b   = (unsigned short*)carve((size_t)HH * FEA * 2);
    unsigned short* W2b   = (unsigned short*)carve((size_t)HH * HH * 2);
    unsigned short* Wihb  = (unsigned short*)carve((size_t)HH * HH * 2);
    unsigned short* Whhb  = (unsigned short*)carve((size_t)HH * HH * 2);
    unsigned short* Wsumb = (unsigned short*)carve((size_t)HH * HH * 2);
    unsigned short* x1b   = (unsigned short*)carve((size_t)BB * HH * 2);
    unsigned short* xlowb = (unsigned short*)carve((size_t)BB * HH * 2);
    unsigned short* hxbA  = (unsigned short*)carve((size_t)BB * HH * 2);
    unsigned short* hxbB  = (unsigned short*)carve((size_t)BB * HH * 2);
    float*          parts = (float*)carve((size_t)TT * BB * 16 * 4);  // actor partials
    float*          biasr = (float*)carve((size_t)HH * 4);

    // --- precision prep: fp32 -> bf16 (RNE); Wsum = Wih+Whh; bias = bih+bhh ---
    int n;
    n = BB * FEA;  k_f32_to_bf16<<<(n + 255) / 256, 256, 0, stream>>>(x,   Xb,   n);
    n = HH * FEA;  k_f32_to_bf16<<<(n + 255) / 256, 256, 0, stream>>>(W1,  W1b,  n);
    n = HH * HH;   k_f32_to_bf16<<<(n + 255) / 256, 256, 0, stream>>>(W2,  W2b,  n);
    n = HH * HH;   k_f32_to_bf16<<<(n + 255) / 256, 256, 0, stream>>>(Wih, Wihb, n);
    n = HH * HH;   k_f32_to_bf16<<<(n + 255) / 256, 256, 0, stream>>>(Whh, Whhb, n);
    n = HH * HH;   k_sum_to_bf16<<<(n + 255) / 256, 256, 0, stream>>>(Wih, Whh, Wsumb, n);
    n = HH;        k_add_f32<<<(n + 255) / 256, 256, 0, stream>>>(bih, bhh, biasr, n);

    // --- grid: 256 tiles of 64x64, 2 tiles per block, 2-way split-K => 512 waves ---
    const int tilesN = HH / 64;                        // 16
    const int tiles  = (BB / 64) * tilesN;             // 256
    dim3 gg(tiles / 2), gb(128);

    // --- low_fc: two GEMMs with fused ReLU ---
    k_gemm_wmma<<<gg, gb, 0, stream>>>(Xb,  W1b, nullptr, nullptr, b1, nullptr, x1b,
                                       nullptr, nullptr, BB, HH, FEA, /*relu*/1, tilesN);
    k_gemm_wmma<<<gg, gb, 0, stream>>>(x1b, W2b, nullptr, nullptr, b2, nullptr, xlowb,
                                       nullptr, nullptr, BB, HH, HH, /*relu*/1, tilesN);

    // --- RNN scan: hx(0) = x_low.
    // t%GAP==0 && t>0 -> x_low@Wih^T + hx@Whh^T ; else (inp==hx) -> hx@(Wih+Whh)^T.
    // Actor head fused: per-step per-row partial dots land in `parts`.
    const unsigned short* hcur = xlowb;
    unsigned short* hnext = hxbA;
    for (int t = 0; t < TT; ++t) {
        float* of = (t == TT - 1) ? (out + (size_t)TT * BB) : nullptr;  // last step -> d_out hx
        float* pp = parts + (size_t)t * BB * 16;
        if (t > 0 && (t % GAP) == 0) {
            k_gemm_wmma<<<gg, gb, 0, stream>>>(xlowb, Wihb, hcur, Whhb, biasr, of, hnext,
                                               Wa, pp, BB, HH, HH, /*tanh*/2, tilesN);
        } else {
            k_gemm_wmma<<<gg, gb, 0, stream>>>(hcur, Wsumb, nullptr, nullptr, biasr, of, hnext,
                                               Wa, pp, BB, HH, HH, /*tanh*/2, tilesN);
        }
        hcur  = hnext;
        hnext = (hnext == hxbA) ? hxbB : hxbA;
    }

    // --- one fused sigmoid pass over all T*B logits ---
    n = TT * BB;
    k_sigmoid_all<<<(n + 255) / 256, 256, 0, stream>>>(parts, ba, out, n, 16);
}
